// CROMAMBA_18090402250978
// MI455X (gfx1250) — matrix operand order repaired and checked by reading, not compiled, not run on hardware
//
#include <hip/hip_runtime.h>
#include <hip/hip_bf16.h>
#include <math.h>

// ---------------- problem constants ----------------
#define B_SZ 128
#define DI   64
#define KD   4
#define NS   24      // d_state
#define RR   24      // dt_rank
#define LL   256
#define C72  72      // R + 2N
#define EPSV 1e-5f

typedef __attribute__((ext_vector_type(16))) _Float16 v16h;
typedef __attribute__((ext_vector_type(8)))  float    v8f;

__device__ __forceinline__ int laneid() { return (int)(threadIdx.x & 31u); }

__device__ __forceinline__ float fast_rcp(float x) { return __builtin_amdgcn_rcpf(x); }
__device__ __forceinline__ float silu(float v) { return v * fast_rcp(1.0f + __expf(-v)); }
__device__ __forceinline__ float lane_bcast(float v, int n) {
    return __int_as_float(__builtin_amdgcn_readlane(__float_as_int(v), n));
}

__device__ __forceinline__ v8f wmma_f16(v16h a, v16h b, v8f c) {
    // D = A(16x32 f16) * B(32x16 f16) + C(16x16 f32)
    return __builtin_amdgcn_wmma_f32_16x16x32_f16(
        /*neg_a=*/false, a, /*neg_b=*/false, b,
        /*c_mod=*/(short)0, c, /*reuse_a=*/false, /*reuse_b=*/false);
}

// error-corrected f16 fragment pair: x ~= (float)hi + (float)lo
struct fragp { v16h hi, lo; };
struct h2 { _Float16 hi, lo; };

__device__ __forceinline__ h2 split_f32(float x) {
    h2 r;
    r.hi = (_Float16)x;
    r.lo = (_Float16)(x - (float)r.hi);
    return r;
}

// D += A*B with Ootomo error correction: Alo*Bhi + Ahi*Blo + Ahi*Bhi
__device__ __forceinline__ v8f wmma_ec(const fragp& a, const fragp& b, v8f acc) {
    acc = wmma_f16(a.lo, b.hi, acc);
    acc = wmma_f16(a.hi, b.lo, acc);
    acc = wmma_f16(a.hi, b.hi, acc);
    return acc;
}

// ---- fragment loaders (ISA 7.12.2 layouts), f32 memory -> split f16 regs ----
// A 16x32: lane m = m0 + (lane&15); half = lane>>4; elem e -> K = (e>>3)*16 + half*8 + (e&7)
__device__ __forceinline__ fragp frag_a_rm(const float* __restrict__ src, int ld, int m0, int k0) {
    const int lane = laneid();
    const int m  = m0 + (lane & 15);
    const int kb = k0 + ((lane >> 4) << 3);
    fragp a;
#pragma unroll
    for (int e = 0; e < 16; ++e) {
        const int kk = kb + ((e >> 3) << 4) + (e & 7);
        const h2 s = split_f32(src[m * ld + kk]);
        a.hi[e] = s.hi; a.lo[e] = s.lo;
    }
    return a;
}
// B 32x16 row-major B[K,N] = src[k*ld + n]
__device__ __forceinline__ fragp frag_b_rm(const float* __restrict__ src, int ld, int k0, int n0) {
    const int lane = laneid();
    const int n  = n0 + (lane & 15);
    const int kb = k0 + ((lane >> 4) << 3);
    fragp b;
#pragma unroll
    for (int e = 0; e < 16; ++e) {
        const h2 s = split_f32(src[(kb + ((e >> 3) << 4) + (e & 7)) * ld + n]);
        b.hi[e] = s.hi; b.lo[e] = s.lo;
    }
    return b;
}
// B 32x16 from transposed weight W(N,K) row-major: B[K,N] = W[n*ld + k]
__device__ __forceinline__ fragp frag_b_tr(const float* __restrict__ src, int ld, int k0, int n0) {
    const int lane = laneid();
    const int n  = n0 + (lane & 15);
    const int kb = k0 + ((lane >> 4) << 3);
    fragp b;
#pragma unroll
    for (int e = 0; e < 16; ++e) {
        const h2 s = split_f32(src[n * ld + kb + ((e >> 3) << 4) + (e & 7)]);
        b.hi[e] = s.hi; b.lo[e] = s.lo;
    }
    return b;
}

// scan-direction permutations: xs[b,k,d,l] = x[b,d,sigma(k,l)]
__device__ __forceinline__ int hw_tr(int l) { return ((l & 15) << 4) | (l >> 4); }
__device__ __forceinline__ int sigma(int k, int l) {
    switch (k & 3) {
        case 0: return l;
        case 1: return hw_tr(l);
        case 2: return 255 - l;
        default: return 255 - hw_tr(l);
    }
}
__device__ __forceinline__ int sigma_inv(int k, int l) {
    switch (k & 3) {
        case 0: return l;
        case 1: return hw_tr(l);
        case 2: return 255 - l;
        default: return hw_tr(255 - l);
    }
}

// ============ K1: z = silu(X @ W^T), X:(32768,64) W:(64,64) ============
__global__ void __launch_bounds__(256) k_inproj_silu(const float* __restrict__ X,
                                                     const float* __restrict__ W,
                                                     float* __restrict__ Z) {
    const int wave = blockIdx.x * 8 + (threadIdx.x >> 5);
    const int mt = wave >> 2, nt = wave & 3;
    const int m0 = mt << 4, n0 = nt << 4;
    v8f acc = {};
    acc = wmma_ec(frag_a_rm(X, DI, m0, 0),  frag_b_tr(W, DI, 0,  n0), acc);
    acc = wmma_ec(frag_a_rm(X, DI, m0, 32), frag_b_tr(W, DI, 32, n0), acc);
    const int lane = laneid();
    const int n  = n0 + (lane & 15);
    const int mb = m0 + ((lane >> 4) << 3);
#pragma unroll
    for (int r = 0; r < 8; ++r) {
        Z[(size_t)(mb + r) * DI + n] = silu(acc[r]);
    }
}

// ============ K2: pixel-unshuffle + depthwise 3x3 + bias + silu ============
// Xin:(b,16,16,64) -> XC:(b,4,64,64)
__global__ void __launch_bounds__(256) k_dwconv(const float* __restrict__ Xin,
                                                const float* __restrict__ Wc,
                                                const float* __restrict__ bc,
                                                float* __restrict__ XC) {
    const int idx = blockIdx.x * blockDim.x + threadIdx.x;   // B*4*64*64
    const int x = idx & 63, y = (idx >> 6) & 63, c = (idx >> 12) & 3, b = idx >> 14;
    float acc = 0.0f;
#pragma unroll
    for (int dy = 0; dy < 3; ++dy) {
        const int yy = y + dy - 1;
#pragma unroll
        for (int dx = 0; dx < 3; ++dx) {
            const int xx = x + dx - 1;
            float v = 0.0f;
            if (yy >= 0 && yy < 64 && xx >= 0 && xx < 64) {
                // unshuffle: in[b,c,yy,xx] = Xin[b, yy/4, xx/4, c*16 + (yy%4)*4 + (xx%4)]
                v = Xin[(((size_t)b * 16 + (yy >> 2)) * 16 + (xx >> 2)) * 64
                        + c * 16 + (yy & 3) * 4 + (xx & 3)];
            }
            acc += Wc[c * 9 + dy * 3 + dx] * v;
        }
    }
    acc += bc[c];
    XC[(size_t)idx] = silu(acc);
}

// ============ K3: 4x4 stride-4 patch conv (im2col WMMA) + bias + BN ============
// XC:(b,4,64,64), PW:(64,4,4,4) -> Xout:(b, d, l) with l = h*16+w
__global__ void __launch_bounds__(256) k_patch(const float* __restrict__ XC,
                                               const float* __restrict__ PW,
                                               const float* __restrict__ PB,
                                               const float* __restrict__ BG,
                                               const float* __restrict__ BB,
                                               float* __restrict__ Xout) {
    const int wave = blockIdx.x * 8 + (threadIdx.x >> 5);
    const int mt = wave >> 2, nt = wave & 3;
    const int m0 = mt << 4, n0 = nt << 4;
    const int lane = laneid();
    const int m  = m0 + (lane & 15);               // token row = b*256 + h*16 + w
    const int b  = m >> 8, hw = m & 255;
    const int h4 = (hw >> 4) << 2, w4 = (hw & 15) << 2;
    const int kbase = (lane >> 4) << 3;
    v8f acc = {};
#pragma unroll
    for (int ks = 0; ks < 2; ++ks) {
        fragp a;
#pragma unroll
        for (int e = 0; e < 16; ++e) {
            const int kk = ks * 32 + kbase + ((e >> 3) << 4) + (e & 7);
            const int ci = kk >> 4, dy = (kk >> 2) & 3, dx = kk & 3;
            const h2 s = split_f32(XC[(((size_t)b * 4 + ci) * 64 + h4 + dy) * 64 + w4 + dx]);
            a.hi[e] = s.hi; a.lo[e] = s.lo;
        }
        acc = wmma_ec(a, frag_b_tr(PW, 64, ks * 32, n0), acc);
    }
    const int n  = n0 + (lane & 15);
    const int mb = m0 + ((lane >> 4) << 3);
    const float scale = BG[n] * rsqrtf(1.0f + EPSV);
    const float pbn = PB[n], bbn = BB[n];
#pragma unroll
    for (int r = 0; r < 8; ++r) {
        const int mm = mb + r;
        const float v = (acc[r] + pbn) * scale + bbn;
        Xout[(((size_t)(mm >> 8)) * DI + n) * LL + (mm & 255)] = v;
    }
}

// ============ K4: x_dbl[b,k,c,l] = sum_d xs[b,k,d,l] * Wp[k,c,d] ============
// computed on unpermuted x, stored at sigma_inv(k, l). M=72 padded to 80.
__global__ void __launch_bounds__(256) k_xdbl(const float* __restrict__ X,     // (b,d,l)
                                              const float* __restrict__ XPW,   // (4,72,64)
                                              float* __restrict__ XDBL) {      // (b,4,72,256)
    const int wave = blockIdx.x * 8 + (threadIdx.x >> 5);   // 40960 waves
    const int nt = wave & 15;
    const int mt = (wave >> 4) % 5;
    const int bk = wave / 80;
    const int b = bk >> 2, k = bk & 3;
    const int m0 = mt << 4, n0 = nt << 4;
    const float* __restrict__ W  = XPW + (size_t)k * C72 * DI;
    const float* __restrict__ XB = X + (size_t)b * DI * LL;
    const int lane = laneid();
    const int mc = min(m0 + (lane & 15), C72 - 1);   // clamp pad rows for loads
    const int kbase = (lane >> 4) << 3;
    v8f acc = {};
#pragma unroll
    for (int ks = 0; ks < 2; ++ks) {
        fragp a;
#pragma unroll
        for (int e = 0; e < 16; ++e) {
            const int kk = ks * 32 + kbase + ((e >> 3) << 4) + (e & 7);
            const h2 s = split_f32(W[mc * DI + kk]);
            a.hi[e] = s.hi; a.lo[e] = s.lo;
        }
        acc = wmma_ec(a, frag_b_rm(XB, LL, ks * 32, n0), acc);
    }
    const int n  = n0 + (lane & 15);
    const int lp = sigma_inv(k, n);
    const int mb = m0 + ((lane >> 4) << 3);
    float* __restrict__ dst = XDBL + (size_t)bk * C72 * LL;
#pragma unroll
    for (int r = 0; r < 8; ++r) {
        const int c = mb + r;
        if (c < C72) dst[(size_t)c * LL + lp] = acc[r];
    }
}

// ============ K5: dts[b,k,d,l] = sum_r dt_w[k,d,r] * x_dbl[b,k,r,l] (K=24 pad 32) ====
__global__ void __launch_bounds__(256) k_dts(const float* __restrict__ XDBL,  // (b,4,72,256)
                                             const float* __restrict__ DTW,   // (4,64,24)
                                             float* __restrict__ DTS) {       // (b,4,64,256)
    const int wave = blockIdx.x * 8 + (threadIdx.x >> 5);   // 32768 waves
    const int nt = wave & 15;
    const int mt = (wave >> 4) & 3;
    const int bk = wave >> 6;
    const int k  = bk & 3;
    const int m0 = mt << 4, n0 = nt << 4;
    const float* __restrict__ W  = DTW + (size_t)k * DI * RR;
    const float* __restrict__ XB = XDBL + (size_t)bk * C72 * LL;
    const int lane = laneid();
    const int m = m0 + (lane & 15);
    const int kbase = (lane >> 4) << 3;
    fragp a, bb;
#pragma unroll
    for (int e = 0; e < 16; ++e) {
        const int kk = kbase + ((e >> 3) << 4) + (e & 7);
        const float v = (kk < RR) ? W[m * RR + kk] : 0.0f;
        const h2 s = split_f32(v);
        a.hi[e] = s.hi; a.lo[e] = s.lo;
    }
    const int n  = n0 + (lane & 15);
    const int kb = (lane >> 4) << 3;
#pragma unroll
    for (int e = 0; e < 16; ++e) {
        const int kk = kb + ((e >> 3) << 4) + (e & 7);
        const float v = (kk < RR) ? XB[(size_t)kk * LL + n] : 0.0f;
        const h2 s = split_f32(v);
        bb.hi[e] = s.hi; bb.lo[e] = s.lo;
    }
    v8f acc = {};
    acc = wmma_ec(a, bb, acc);
    const int mb = m0 + ((lane >> 4) << 3);
    float* __restrict__ dst = DTS + (size_t)bk * DI * LL;
#pragma unroll
    for (int r = 0; r < 8; ++r) dst[(size_t)(mb + r) * LL + n] = acc[r];
}

// ============ K6: selective scan ============
// One wave32 per (b,k); lane owns channels d=lane and d=lane+32; B_t/C_t broadcast
// with v_readlane -- no LDS, no barriers in the 256-step serial loop.
__global__ void __launch_bounds__(32) k_scan(const float* __restrict__ Xself,  // (b,d,l)
                                             const float* __restrict__ XDBLo,  // (b,4,72,256)
                                             float* __restrict__ DTSo,         // dts in / y out
                                             const float* __restrict__ AlogO,  // (4,64,24)
                                             const float* __restrict__ DsO,    // (4,64)
                                             const float* __restrict__ dtbO) { // (4,64)
    const int bk = blockIdx.x;           // b*4 + k
    const int b = bk >> 2, k = bk & 3;
    const int lane = laneid();
    const int d0 = lane, d1 = lane + 32;
    float h0[NS], h1[NS], A0[NS], A1[NS];
#pragma unroll
    for (int n = 0; n < NS; ++n) {
        h0[n] = 0.0f; h1[n] = 0.0f;
        A0[n] = -__expf(AlogO[((size_t)k * DI + d0) * NS + n]);
        A1[n] = -__expf(AlogO[((size_t)k * DI + d1) * NS + n]);
    }
    const float Dp0 = DsO[k * DI + d0], Dp1 = DsO[k * DI + d1];
    const float db0 = dtbO[k * DI + d0], db1 = dtbO[k * DI + d1];
    const float* __restrict__ Xr0 = Xself + ((size_t)b * DI + d0) * LL;
    const float* __restrict__ Xr1 = Xself + ((size_t)b * DI + d1) * LL;
    const float* __restrict__ Bb = XDBLo + ((size_t)bk * C72 + RR) * LL;
    const float* __restrict__ Cb = XDBLo + ((size_t)bk * C72 + RR + NS) * LL;
    float* __restrict__ DT0 = DTSo + ((size_t)bk * DI + d0) * LL;
    float* __restrict__ DT1 = DTSo + ((size_t)bk * DI + d1) * LL;
    for (int l = 0; l < LL; ++l) {
        float vB = 0.0f, vC = 0.0f;
        if (lane < NS) {                 // lanes 0..23 fetch B_t and C_t
            vB = Bb[(size_t)lane * LL + l];
            vC = Cb[(size_t)lane * LL + l];
        }
        const int ls = sigma(k, l);
        const float u0 = Xr0[ls], u1 = Xr1[ls];
        const float dv0 = DT0[l] + db0;
        const float dv1 = DT1[l] + db1;
        const float delta0 = (dv0 > 20.0f) ? dv0 : __logf(1.0f + __expf(dv0));
        const float delta1 = (dv1 > 20.0f) ? dv1 : __logf(1.0f + __expf(dv1));
        const float du0 = delta0 * u0, du1 = delta1 * u1;
        float y0 = 0.0f, y1 = 0.0f;
#pragma unroll
        for (int n = 0; n < NS; ++n) {
            const float bn = lane_bcast(vB, n);   // v_readlane broadcast
            const float cn = lane_bcast(vC, n);
            h0[n] = __expf(delta0 * A0[n]) * h0[n] + du0 * bn;
            h1[n] = __expf(delta1 * A1[n]) * h1[n] + du1 * bn;
            y0 += h0[n] * cn;
            y1 += h1[n] * cn;
        }
        DT0[l] = y0 + u0 * Dp0;
        DT1[l] = y1 + u1 * Dp1;
    }
}

// ============ K7a: cross-merge + LayerNorm(DI) + z-gate (in-place over z) ============
__global__ void __launch_bounds__(256) k_merge_ln(const float* __restrict__ Y,   // (b,4,d,l)
                                                  const float* __restrict__ lng,
                                                  const float* __restrict__ lnb,
                                                  float* __restrict__ Z) {       // (token,d)
    const int token = blockIdx.x * 8 + (threadIdx.x >> 5);   // b*256 + l
    const int b = token >> 8, l = token & 255;
    const int lane = laneid();
    const int t = hw_tr(l);
    const float* __restrict__ Yb = Y + (size_t)b * 4 * DI * LL;
    float v[2];
#pragma unroll
    for (int j = 0; j < 2; ++j) {
        const int d = lane + j * 32;
        v[j] = Yb[((size_t)(0 * DI + d)) * LL + l]
             + Yb[((size_t)(2 * DI + d)) * LL + (255 - l)]
             + Yb[((size_t)(1 * DI + d)) * LL + t]
             + Yb[((size_t)(3 * DI + d)) * LL + (255 - t)];
    }
    float s  = v[0] + v[1];
    float sq = v[0] * v[0] + v[1] * v[1];
#pragma unroll
    for (int off = 16; off > 0; off >>= 1) {
        s  += __shfl_xor(s, off, 32);
        sq += __shfl_xor(sq, off, 32);
    }
    const float mu   = s * (1.0f / 64.0f);
    const float var  = sq * (1.0f / 64.0f) - mu * mu;
    const float rstd = rsqrtf(var + EPSV);
#pragma unroll
    for (int j = 0; j < 2; ++j) {
        const int d = lane + j * 32;
        const float g = (v[j] - mu) * rstd * lng[d] + lnb[d];
        float* zp = Z + (size_t)token * DI + d;
        *zp = g * (*zp);
    }
}

// ============ K7b: out = G @ W^T ============
__global__ void __launch_bounds__(256) k_out_gemm(const float* __restrict__ G,
                                                  const float* __restrict__ W,
                                                  float* __restrict__ O) {
    const int wave = blockIdx.x * 8 + (threadIdx.x >> 5);
    const int mt = wave >> 2, nt = wave & 3;
    const int m0 = mt << 4, n0 = nt << 4;
    v8f acc = {};
    acc = wmma_ec(frag_a_rm(G, DI, m0, 0),  frag_b_tr(W, DI, 0,  n0), acc);
    acc = wmma_ec(frag_a_rm(G, DI, m0, 32), frag_b_tr(W, DI, 32, n0), acc);
    const int lane = laneid();
    const int n  = n0 + (lane & 15);
    const int mb = m0 + ((lane >> 4) << 3);
#pragma unroll
    for (int r = 0; r < 8; ++r) O[(size_t)(mb + r) * DI + n] = acc[r];
}

// ============================ launcher ============================
extern "C" void kernel_launch(void* const* d_in, const int* in_sizes, int n_in,
                              void* d_out, int out_size, void* d_ws, size_t ws_size,
                              hipStream_t stream) {
    (void)in_sizes; (void)n_in; (void)out_size; (void)ws_size;
    const float* Xin[2] = { (const float*)d_in[0], (const float*)d_in[1] };
    const float *ipw[2], *cw[2], *cb[2], *pw[2], *pb[2], *bg[2], *bbn[2],
                *xpw[2], *dtw[2], *dtb[2], *alog[2], *dsp[2], *opw[2];
    for (int br = 0; br < 2; ++br) {
        const int o = 2 + br * 13;
        ipw[br]  = (const float*)d_in[o + 0];
        cw[br]   = (const float*)d_in[o + 1];
        cb[br]   = (const float*)d_in[o + 2];
        pw[br]   = (const float*)d_in[o + 3];
        pb[br]   = (const float*)d_in[o + 4];
        bg[br]   = (const float*)d_in[o + 5];
        bbn[br]  = (const float*)d_in[o + 6];
        xpw[br]  = (const float*)d_in[o + 7];
        dtw[br]  = (const float*)d_in[o + 8];
        dtb[br]  = (const float*)d_in[o + 9];
        alog[br] = (const float*)d_in[o + 10];
        dsp[br]  = (const float*)d_in[o + 11];
        opw[br]  = (const float*)d_in[o + 12];
    }
    const float* lng = (const float*)d_in[28];
    const float* lnb = (const float*)d_in[29];

    // workspace partition (floats)
    float* ws = (float*)d_ws;
    const size_t TOK = (size_t)B_SZ * LL * DI;          // 2,097,152
    const size_t XDB = (size_t)B_SZ * KD * C72 * LL;    // 9,437,184
    const size_t DTN = (size_t)B_SZ * KD * DI * LL;     // 8,388,608
    size_t off = 0;
    float *Z[2], *XC[2], *XP[2], *XDBL[2], *DTS[2];
    for (int br = 0; br < 2; ++br) { Z[br]    = ws + off; off += TOK; }
    for (int br = 0; br < 2; ++br) { XC[br]   = ws + off; off += TOK; }
    for (int br = 0; br < 2; ++br) { XP[br]   = ws + off; off += TOK; }
    for (int br = 0; br < 2; ++br) { XDBL[br] = ws + off; off += XDB; }
    for (int br = 0; br < 2; ++br) { DTS[br]  = ws + off; off += DTN; }

    for (int br = 0; br < 2; ++br) {
        k_inproj_silu<<<1024, 256, 0, stream>>>(Xin[br], ipw[br], Z[br]);
        k_dwconv<<<8192, 256, 0, stream>>>(Xin[br], cw[br], cb[br], XC[br]);
        k_patch<<<1024, 256, 0, stream>>>(XC[br], pw[br], pb[br], bg[br], bbn[br], XP[br]);
        k_xdbl<<<5120, 256, 0, stream>>>(XP[br], xpw[br], XDBL[br]);
        k_dts<<<4096, 256, 0, stream>>>(XDBL[br], dtw[br], DTS[br]);
    }
    // cross-modal scans: y_ms = scan(u=MS, params=PAN) -> in-place over DTS[1]
    k_scan<<<512, 32, 0, stream>>>(XP[0], XDBL[1], DTS[1], alog[1], dsp[1], dtb[1]);
    // y_pan = scan(u=PAN, params=MS) -> in-place over DTS[0]
    k_scan<<<512, 32, 0, stream>>>(XP[1], XDBL[0], DTS[0], alog[0], dsp[0], dtb[0]);

    // merge + LN + gate (in-place over z), then out_proj
    k_merge_ln<<<4096, 256, 0, stream>>>(DTS[1], lng, lnb, Z[0]);
    k_merge_ln<<<4096, 256, 0, stream>>>(DTS[0], lng, lnb, Z[1]);

    float* out = (float*)d_out;
    k_out_gemm<<<1024, 256, 0, stream>>>(Z[0], opw[0], out);
    k_out_gemm<<<1024, 256, 0, stream>>>(Z[1], opw[1], out + TOK);
}
